// ResidualVQ_12506944766262
// MI455X (gfx1250) — compile-verified, matrix operand order checked by reading, not run on hardware
//
#include <hip/hip_runtime.h>

// ---------------------------------------------------------------------------
// Residual VQ for MI455X (gfx1250).
//   distances via v_wmma_f32_16x16x32_bf16, B tiles async-staged into LDS
//   (global_load_async_to_lds_b128 + s_wait_asynccnt), residual kept f32 in LDS.
// ---------------------------------------------------------------------------

typedef __attribute__((ext_vector_type(16))) __bf16 v16bf;
typedef __attribute__((ext_vector_type(8)))  float  v8f;

#define BATCH   65536
#define DIM     256
#define KCODES  1024
#define LEVELS  4
#define ROWS_WG 128
#define THREADS 256
#define TILES   (KCODES / 16)              // 64 code-tiles of 16 codes
#define TILE_BYTES 8192                    // 8 kc-chunks * 32 lanes * 32 B
#define PACKED_BYTES ((size_t)LEVELS * TILES * TILE_BYTES)   // 2 MB

static __device__ __forceinline__ unsigned short f2bf(float f) {
  __bf16 b = (__bf16)f;
  unsigned short u;
  __builtin_memcpy(&u, &b, 2);
  return u;
}

// ---------------------------------------------------------------------------
// Pack codebooks (f32) into WMMA-ready bf16 B fragments.
// Fragment layout: [level][tile][kc][lane] -> 16 contiguous bf16 (32 B):
//   lane: n = lane&15 (code within tile), h = lane>>4
//   values j=0..15:  cb[level][tile*16+n][kc*32 + h*16 + j]
// One thread per fragment: 65536 threads total.
// ---------------------------------------------------------------------------
__global__ void vq_pack_kernel(const float* __restrict__ cb,
                               char* __restrict__ packed) {
  int t    = blockIdx.x * blockDim.x + threadIdx.x;   // 0..65535
  int lane = t & 31;
  int kc   = (t >> 5) & 7;
  int tile = (t >> 8) & 63;
  int lvl  = t >> 14;
  int n = lane & 15, h = lane >> 4;
  const float* src = cb + (((size_t)lvl * KCODES) + tile * 16 + n) * DIM
                        + kc * 32 + h * 16;
  unsigned wv[8];
#pragma unroll
  for (int j = 0; j < 8; ++j)
    wv[j] = (unsigned)f2bf(src[2 * j]) | ((unsigned)f2bf(src[2 * j + 1]) << 16);
  uint4* dst = (uint4*)(packed + ((size_t)t << 5));
  dst[0] = make_uint4(wv[0], wv[1], wv[2], wv[3]);
  dst[1] = make_uint4(wv[4], wv[5], wv[6], wv[7]);
}

// ---------------------------------------------------------------------------
// Per-code squared norms ||c||^2 (exact f32). One wave per code row.
// ---------------------------------------------------------------------------
__global__ void vq_norm_kernel(const float* __restrict__ cb,
                               float* __restrict__ cnorm) {
  int wave = threadIdx.x >> 5, lane = threadIdx.x & 31;
  int row  = blockIdx.x * 8 + wave;                   // 512 blocks * 8 waves
  const float* src = cb + (size_t)row * DIM;
  float s = 0.f;
#pragma unroll
  for (int i = 0; i < 8; ++i) { float v = src[lane + i * 32]; s = fmaf(v, v, s); }
#pragma unroll
  for (int w = 16; w > 0; w >>= 1) s += __shfl_xor(s, w, 32);
  if (lane == 0) cnorm[row] = s;
}

// ---------------------------------------------------------------------------
// Main kernel: 512 WGs x 256 threads; each WG owns 128 rows (16 per wave).
// ---------------------------------------------------------------------------
__global__ __launch_bounds__(THREADS) void vq_main_kernel(
    const float* __restrict__ z, const float* __restrict__ cb,
    const char* __restrict__ packed, const float* __restrict__ cnorm,
    float* __restrict__ zq, float* __restrict__ codes_out,
    float* __restrict__ partial) {
  __shared__ __align__(16) float r_s[ROWS_WG * DIM];      // 128 KB residual (f32)
  __shared__ __align__(32) char  bbuf[2 * TILE_BYTES];    // 16 KB double-buffer
  __shared__ int   codes_s[ROWS_WG];
  __shared__ float wsum[8];

  const int tid  = threadIdx.x;
  const int wave = tid >> 5, lane = tid & 31;
  const int m = lane & 15, h = lane >> 4;
  const int rowBase = blockIdx.x * ROWS_WG;

  // ---- load z tile into LDS residual (coalesced float4) ----
  {
    const float4* z4 = (const float4*)z;
    float4* r4 = (float4*)r_s;
    size_t base4 = (size_t)rowBase * (DIM / 4);
#pragma unroll
    for (int i = 0; i < 32; ++i)
      r4[tid + (i << 8)] = z4[base4 + tid + (i << 8)];
  }
  __syncthreads();

  for (int lvl = 0; lvl < LEVELS; ++lvl) {
    // ---- build bf16 A fragments (16 rows x K=256) from residual ----
    v16bf a[8];
    {
      const float* rp0 = &r_s[(((wave << 4) + m) * DIM) + (h << 4)];
#pragma unroll
      for (int kc = 0; kc < 8; ++kc) {
        const float* rp = rp0 + (kc << 5);
#pragma unroll
        for (int j = 0; j < 16; ++j) a[kc][j] = (__bf16)rp[j];
      }
    }

    float best[8];
    int   bidx[8];
#pragma unroll
    for (int v = 0; v < 8; ++v) { best[v] = 3.4e38f; bidx[v] = 0; }

    const char*  packL  = packed + ((size_t)lvl * TILES) * TILE_BYTES;
    const float* cnormL = cnorm + lvl * KCODES;

    // ---- prologue: async-stage tile 0 into bbuf[0] ----
    {
      unsigned dl = (unsigned)(size_t)(void*)bbuf + (unsigned)(tid << 5);
      const char* srcp = packL + (tid << 5);
      asm volatile("global_load_async_to_lds_b128 %0, %1, off"
                   :: "v"(dl), "v"(srcp) : "memory");
      asm volatile("global_load_async_to_lds_b128 %0, %1, off offset:16"
                   :: "v"(dl), "v"(srcp) : "memory");
    }

    for (int t = 0; t < TILES; ++t) {
      if (t + 1 < TILES) {
        unsigned dl = (unsigned)(size_t)(void*)bbuf
                    + (unsigned)(((t + 1) & 1) * TILE_BYTES) + (unsigned)(tid << 5);
        const char* srcp = packL + (size_t)(t + 1) * TILE_BYTES + (tid << 5);
        asm volatile("global_load_async_to_lds_b128 %0, %1, off"
                     :: "v"(dl), "v"(srcp) : "memory");
        asm volatile("global_load_async_to_lds_b128 %0, %1, off offset:16"
                     :: "v"(dl), "v"(srcp) : "memory");
        asm volatile("s_wait_asynccnt 2" ::: "memory");   // tile t arrived
      } else {
        asm volatile("s_wait_asynccnt 0" ::: "memory");
      }
      __syncthreads();   // tile t staged & visible to all waves

      const v16bf* bp = (const v16bf*)(bbuf + (t & 1) * TILE_BYTES);
      float cn = cnormL[(t << 4) + m];
      v8f acc = {};
#pragma unroll
      for (int kc = 0; kc < 8; ++kc) {
        v16bf b = bp[(kc << 5) + lane];
        acc = __builtin_amdgcn_wmma_f32_16x16x32_bf16(
            false, a[kc], false, b, (short)0, acc, false, false);
      }
      const int cbase = (t << 4) + m;
#pragma unroll
      for (int v = 0; v < 8; ++v) {
        float s = fmaf(-2.0f, acc[v], cn);   // ||r||^2 constant per row: dropped
        if (s < best[v]) { best[v] = s; bidx[v] = cbase; }
      }
      __syncthreads();   // everyone done reading bbuf[t&1]
    }

    // ---- per-row argmin across the 16 lanes of each half ----
#pragma unroll
    for (int v = 0; v < 8; ++v) {
#pragma unroll
      for (int w = 1; w < 16; w <<= 1) {
        float os = __shfl_xor(best[v], w, 32);
        int   oi = __shfl_xor(bidx[v], w, 32);
        if (os < best[v] || (os == best[v] && oi < bidx[v])) {
          best[v] = os; bidx[v] = oi;
        }
      }
      if (m == 0) {
        int row = (wave << 4) + v + (h << 3);
        codes_s[row] = bidx[v];
        codes_out[((size_t)(rowBase + row) << 2) + lvl] = (float)bidx[v];
      }
    }
    __syncthreads();

    // ---- residual update: r -= codebook[selected] (f32, L2-resident) ----
    for (int mm = 0; mm < 16; ++mm) {
      int row = (wave << 4) + mm;
      const float* crow = cb + (((size_t)lvl << 10) + codes_s[row]) * DIM;
#pragma unroll
      for (int c8 = 0; c8 < 8; ++c8) {
        int d = (c8 << 5) + lane;
        r_s[row * DIM + d] -= crow[d];
      }
    }
    __syncthreads();
  }

  // ---- commit-loss partial: sum(r_final^2) over this WG's rows ----
  float ls = 0.f;
  for (int i = 0; i < 128; ++i) {
    float e = r_s[(i << 8) + tid];
    ls = fmaf(e, e, ls);
  }
#pragma unroll
  for (int w = 16; w > 0; w >>= 1) ls += __shfl_xor(ls, w, 32);
  if (lane == 0) wsum[wave] = ls;
  __syncthreads();
  if (tid == 0) {
    float s = 0.f;
#pragma unroll
    for (int i = 0; i < 8; ++i) s += wsum[i];
    partial[blockIdx.x] = s;
  }

  // ---- z_q = z - r_final ----
  {
    const float4* z4 = (const float4*)z;
    const float4* r4 = (const float4*)r_s;
    float4* o4 = (float4*)zq;
    size_t base4 = (size_t)rowBase * (DIM / 4);
#pragma unroll
    for (int i = 0; i < 32; ++i) {
      float4 zv = z4[base4 + tid + (i << 8)];
      float4 rv = r4[tid + (i << 8)];
      o4[base4 + tid + (i << 8)] =
          make_float4(zv.x - rv.x, zv.y - rv.y, zv.z - rv.z, zv.w - rv.w);
    }
  }
}

// ---------------------------------------------------------------------------
// Final loss reduction: 512 per-WG partials -> mean over B*D.
// ---------------------------------------------------------------------------
__global__ void vq_loss_reduce(const float* __restrict__ partial,
                               float* __restrict__ out) {
  __shared__ float s[512];
  s[threadIdx.x] = partial[threadIdx.x];
  __syncthreads();
  for (int w = 256; w > 0; w >>= 1) {
    if (threadIdx.x < w) s[threadIdx.x] += s[threadIdx.x + w];
    __syncthreads();
  }
  if (threadIdx.x == 0)
    out[0] = s[0] * (1.0f / ((float)BATCH * (float)DIM));
}

// ---------------------------------------------------------------------------
extern "C" void kernel_launch(void* const* d_in, const int* in_sizes, int n_in,
                              void* d_out, int out_size, void* d_ws, size_t ws_size,
                              hipStream_t stream) {
  (void)in_sizes; (void)n_in; (void)out_size; (void)ws_size;
  const float* z  = (const float*)d_in[0];   // [65536, 256] f32
  const float* cb = (const float*)d_in[1];   // [4, 1024, 256] f32

  float* out   = (float*)d_out;
  float* zq    = out;                                      // 65536*256
  float* codes = out + (size_t)BATCH * DIM;                // 65536*4
  float* loss  = codes + (size_t)BATCH * LEVELS;           // 1

  char*  ws      = (char*)d_ws;
  char*  packed  = ws;                                     // 2 MB bf16 fragments
  float* cnorm   = (float*)(ws + PACKED_BYTES);            // 4096 f32
  float* partial = cnorm + LEVELS * KCODES;                // 512 f32

  vq_pack_kernel<<<(LEVELS * TILES * 8 * 32) / THREADS, THREADS, 0, stream>>>(cb, packed);
  vq_norm_kernel<<<(LEVELS * KCODES) / 8, THREADS, 0, stream>>>(cb, cnorm);
  vq_main_kernel<<<BATCH / ROWS_WG, THREADS, 0, stream>>>(z, cb, packed, cnorm,
                                                          zq, codes, partial);
  vq_loss_reduce<<<1, 512, 0, stream>>>(partial, loss);
}
